// SynthesizerRandomAttentionMergeFeatureMap_70772471103580
// MI455X (gfx1250) — compile-verified, matrix operand order checked by reading, not run on hardware
//
#include <hip/hip_runtime.h>
#include <hip/hip_bf16.h>

// ---- CDNA5 wave32 WMMA types -------------------------------------------------
typedef __attribute__((ext_vector_type(16))) __bf16        v16bf;
typedef __attribute__((ext_vector_type(8)))  float          v8f;
typedef __attribute__((ext_vector_type(8)))  unsigned short v8us;

union Frag {
  v8us  h[2];   // two 16-byte halves (K 0..7+kb, K 16..23+kb per lane)
  v16bf b;      // view consumed by v_wmma_f32_16x16x32_bf16
};

__device__ __forceinline__ unsigned short f2bf(float f) {
  unsigned int u = __builtin_bit_cast(unsigned int, f);
  u += 0x7FFFu + ((u >> 16) & 1u);          // round-to-nearest-even
  return (unsigned short)(u >> 16);
}

// base uniform, off per-lane 32-bit: lowers to SADDR + 32-bit VGPR offset.
__device__ __forceinline__ void load_frag(Frag& f, const unsigned short* base, int off) {
  f.h[0] = *(const v8us*)(base + off);
  f.h[1] = *(const v8us*)(base + off + 16);
}

#define WMMA_BF16(A, B, C) \
  __builtin_amdgcn_wmma_f32_16x16x32_bf16(false, (A), false, (B), (short)0, (C), false, false)

// ---- prep: concat + fp32->bf16, dual layout ([c,n] and [n,c]) ----------------
__global__ void prep_x_kernel(const float* __restrict__ x1, const float* __restrict__ x2,
                              unsigned short* __restrict__ xcn, unsigned short* __restrict__ xnc) {
  int idx = blockIdx.x * 256 + threadIdx.x;          // 16*512*1024 elements
  int n = idx & 1023;
  int c = (idx >> 10) & 511;
  int b = idx >> 19;
  float v = (c < 256) ? x1[(b * 256 + c) * 1024 + n]
                      : x2[(b * 256 + (c - 256)) * 1024 + n];
  unsigned short us = f2bf(v);
  xcn[idx] = us;                                     // [b][c][n]
  xnc[((size_t)b << 19) + n * 512 + c] = us;         // [b][n][c]
}

// ---- prep: 3x3 weights -> bf16, tap-major [tap][co][ci] ----------------------
__global__ void prep_gw_kernel(const float* __restrict__ gw, unsigned short* __restrict__ wpack) {
  int idx = blockIdx.x * 256 + threadIdx.x;          // 512*512*9
  int tap  = idx % 9;
  int rest = idx / 9;
  int ci = rest & 511;
  int co = rest >> 9;
  wpack[tap * 262144 + co * 512 + ci] = f2bf(gw[idx]);
}

__global__ void prep_ww_kernel(const float* __restrict__ ww, unsigned short* __restrict__ wwb) {
  int idx = blockIdx.x * 256 + threadIdx.x;          // 256*512
  wwb[idx] = f2bf(ww[idx]);
}

__global__ void zero_pad_kernel(unsigned short* __restrict__ zp) {
  zp[blockIdx.x * 256 + threadIdx.x] = 0;            // 512 ushorts (1 KB)
}

// ---- random_map = weight1 @ weight2^T (tiny K=16, fp32) ----------------------
__global__ void rmap_kernel(const float* __restrict__ w1, const float* __restrict__ w2,
                            float* __restrict__ rmap) {
  int idx = blockIdx.x * 256 + threadIdx.x;          // 512*512
  int d = idx & 511, c = idx >> 9;
  float s = 0.f;
  #pragma unroll
  for (int j = 0; j < 16; ++j) s += w1[c * 16 + j] * w2[d * 16 + j];
  rmap[idx] = s;
}

// ---- 3x3 SAME conv as 9-tap implicit GEMM; 32x64 tile per wave ---------------
// Zero-padding: OOB lanes redirect their offset into a zeroed pad row, so all
// loads are unconditional (no EXEC masking in the hot loop).
__global__ void conv3x3_wmma_kernel(const unsigned short* __restrict__ xnc,
                                    const unsigned short* __restrict__ wpack,
                                    unsigned short* __restrict__ valt,
                                    const int zoff) {
  const int lane = threadIdx.x & 31;
  const int wave = threadIdx.x >> 5;
  const int n0 = (blockIdx.x * 8 + wave) * 64;       // spatial tile (64 wide)
  const int m0 = blockIdx.y * 32;                    // out-channel tile (32 tall)
  const int b  = blockIdx.z;
  const int kb  = (lane >> 4) * 8;
  const int l15 = lane & 15;
  const int bb0 = b << 19;                           // batch offset (elements)

  int nl[4], py[4], px[4];
  #pragma unroll
  for (int j = 0; j < 4; ++j) {
    nl[j] = n0 + j * 16 + l15;
    py[j] = nl[j] >> 5;
    px[j] = nl[j] & 31;
  }

  v8f acc[2][4] = {};
  #pragma unroll
  for (int tap = 0; tap < 9; ++tap) {                // unrolled: dy/dx literal
    const int dy = tap / 3 - 1, dx = tap % 3 - 1;
    int boff[4];
    #pragma unroll
    for (int j = 0; j < 4; ++j) {
      int yy = py[j] + dy, xx = px[j] + dx;
      bool valid = ((unsigned)yy < 32u) && ((unsigned)xx < 32u);
      boff[j] = valid ? (bb0 + (yy * 32 + xx) * 512 + kb) : (zoff + kb);
    }
    const int aoff0 = tap * 262144 + (m0 + l15) * 512 + kb;
    const int aoff1 = aoff0 + 16 * 512;
    #pragma unroll 2
    for (int k0 = 0; k0 < 512; k0 += 32) {
      Frag fa0, fa1, fb[4];
      load_frag(fa0, wpack, aoff0 + k0);
      load_frag(fa1, wpack, aoff1 + k0);
      #pragma unroll
      for (int j = 0; j < 4; ++j) load_frag(fb[j], xnc, boff[j] + k0);
      #pragma unroll
      for (int j = 0; j < 4; ++j) {
        acc[0][j] = WMMA_BF16(fa0.b, fb[j].b, acc[0][j]);
        acc[1][j] = WMMA_BF16(fa1.b, fb[j].b, acc[1][j]);
      }
    }
  }
  const int half = lane >> 4;
  #pragma unroll
  for (int i = 0; i < 2; ++i)
    #pragma unroll
    for (int j = 0; j < 4; ++j) {
      v8us pack;
      #pragma unroll
      for (int r = 0; r < 8; ++r) pack[r] = f2bf(acc[i][j][r]);
      *(v8us*)(valt + (size_t)(bb0 + nl[j] * 512 + m0 + i * 16 + half * 8)) = pack;
    }
}

// ---- sim = (q @ q^T) * c^-0.5 + random_map; 32x64 tile per wave --------------
__global__ void sim_wmma_kernel(const unsigned short* __restrict__ xcn,
                                const float* __restrict__ rmap,
                                float* __restrict__ simb) {
  const int lane = threadIdx.x & 31;
  const int wave = threadIdx.x >> 5;
  const int n0 = wave * 64;                          // d (512 = 8 waves * 64)
  const int m0 = blockIdx.y * 32;                    // c
  const int b  = blockIdx.z;
  const int kb  = (lane >> 4) * 8;
  const int l15 = lane & 15;
  const int bb0 = b << 19;
  const int aoff0 = bb0 + (m0 + l15) * 1024 + kb;
  const int aoff1 = aoff0 + 16 * 1024;
  int boff[4];
  #pragma unroll
  for (int j = 0; j < 4; ++j) boff[j] = bb0 + (n0 + j * 16 + l15) * 1024 + kb;

  v8f acc[2][4] = {};
  #pragma unroll 2
  for (int k0 = 0; k0 < 1024; k0 += 32) {
    Frag fa0, fa1, fb[4];
    load_frag(fa0, xcn, aoff0 + k0);
    load_frag(fa1, xcn, aoff1 + k0);
    #pragma unroll
    for (int j = 0; j < 4; ++j) load_frag(fb[j], xcn, boff[j] + k0);
    #pragma unroll
    for (int j = 0; j < 4; ++j) {
      acc[0][j] = WMMA_BF16(fa0.b, fb[j].b, acc[0][j]);
      acc[1][j] = WMMA_BF16(fa1.b, fb[j].b, acc[1][j]);
    }
  }
  const float scale = 0.044194173824159216f;         // 512^-0.5
  const int half = lane >> 4;
  float* srow = simb + (size_t)b * 262144;
  #pragma unroll
  for (int i = 0; i < 2; ++i)
    #pragma unroll
    for (int j = 0; j < 4; ++j) {
      int d = n0 + j * 16 + l15;
      #pragma unroll
      for (int r = 0; r < 8; ++r) {
        int c = m0 + i * 16 + half * 8 + r;
        srow[c * 512 + d] = acc[i][j][r] * scale + rmap[c * 512 + d];
      }
    }
}

// ---- row softmax over 512, emit bf16 attn ------------------------------------
__global__ void softmax_kernel(const float* __restrict__ simb, unsigned short* __restrict__ attn) {
  __shared__ float red[256];
  const int tid = threadIdx.x;
  const size_t row = (size_t)blockIdx.x * 512;       // blockIdx.x = b*512 + c
  float v0 = simb[row + tid];
  float v1 = simb[row + tid + 256];
  red[tid] = fmaxf(v0, v1);
  __syncthreads();
  for (int s = 128; s > 0; s >>= 1) {
    if (tid < s) red[tid] = fmaxf(red[tid], red[tid + s]);
    __syncthreads();
  }
  const float m = red[0];
  __syncthreads();
  float e0 = __expf(v0 - m), e1 = __expf(v1 - m);
  red[tid] = e0 + e1;
  __syncthreads();
  for (int s = 128; s > 0; s >>= 1) {
    if (tid < s) red[tid] += red[tid + s];
    __syncthreads();
  }
  const float inv = 1.0f / red[0];
  attn[row + tid]       = f2bf(e0 * inv);
  attn[row + tid + 256] = f2bf(e1 * inv);
}

// ---- ctx = attn @ value; 32x64 tile per wave; stored transposed [n][c] -------
__global__ void ctx_wmma_kernel(const unsigned short* __restrict__ attn,
                                const unsigned short* __restrict__ valt,
                                unsigned short* __restrict__ ctxt) {
  const int lane = threadIdx.x & 31;
  const int wave = threadIdx.x >> 5;
  const int n0 = (blockIdx.x * 8 + wave) * 64;       // spatial n
  const int m0 = blockIdx.y * 32;                    // channel c
  const int b  = blockIdx.z;
  const int kb  = (lane >> 4) * 8;
  const int l15 = lane & 15;
  const int aoff0 = b * 262144 + (m0 + l15) * 512 + kb;
  const int aoff1 = aoff0 + 16 * 512;
  const int bb0 = b << 19;
  int boff[4];
  #pragma unroll
  for (int j = 0; j < 4; ++j) boff[j] = bb0 + (n0 + j * 16 + l15) * 512 + kb;

  v8f acc[2][4] = {};
  #pragma unroll 2
  for (int k0 = 0; k0 < 512; k0 += 32) {
    Frag fa0, fa1, fb[4];
    load_frag(fa0, attn, aoff0 + k0);
    load_frag(fa1, attn, aoff1 + k0);
    #pragma unroll
    for (int j = 0; j < 4; ++j) load_frag(fb[j], valt, boff[j] + k0);
    #pragma unroll
    for (int j = 0; j < 4; ++j) {
      acc[0][j] = WMMA_BF16(fa0.b, fb[j].b, acc[0][j]);
      acc[1][j] = WMMA_BF16(fa1.b, fb[j].b, acc[1][j]);
    }
  }
  const int half = lane >> 4;
  #pragma unroll
  for (int i = 0; i < 2; ++i)
    #pragma unroll
    for (int j = 0; j < 4; ++j) {
      v8us pack;
      #pragma unroll
      for (int r = 0; r < 8; ++r) pack[r] = f2bf(acc[i][j][r]);
      *(v8us*)(ctxt + (size_t)(bb0 + (n0 + j * 16 + l15) * 512 + m0 + i * 16 + half * 8)) = pack;
    }
}

// ---- out = w_w(1x1) @ ctx; 32x64 tile per wave; fp32 NCHW out ----------------
__global__ void out_wmma_kernel(const unsigned short* __restrict__ wwb,
                                const unsigned short* __restrict__ ctxt,
                                float* __restrict__ out) {
  const int lane = threadIdx.x & 31;
  const int wave = threadIdx.x >> 5;
  const int n0 = (blockIdx.x * 8 + wave) * 64;       // spatial
  const int m0 = blockIdx.y * 32;                    // out channel (256)
  const int b  = blockIdx.z;
  const int kb  = (lane >> 4) * 8;
  const int l15 = lane & 15;
  const int aoff0 = (m0 + l15) * 512 + kb;
  const int aoff1 = aoff0 + 16 * 512;
  const int bb0 = b << 19;
  int boff[4];
  #pragma unroll
  for (int j = 0; j < 4; ++j) boff[j] = bb0 + (n0 + j * 16 + l15) * 512 + kb;

  v8f acc[2][4] = {};
  #pragma unroll 2
  for (int k0 = 0; k0 < 512; k0 += 32) {
    Frag fa0, fa1, fb[4];
    load_frag(fa0, wwb, aoff0 + k0);
    load_frag(fa1, wwb, aoff1 + k0);
    #pragma unroll
    for (int j = 0; j < 4; ++j) load_frag(fb[j], ctxt, boff[j] + k0);
    #pragma unroll
    for (int j = 0; j < 4; ++j) {
      acc[0][j] = WMMA_BF16(fa0.b, fb[j].b, acc[0][j]);
      acc[1][j] = WMMA_BF16(fa1.b, fb[j].b, acc[1][j]);
    }
  }
  const int half = lane >> 4;
  float* orow = out + (size_t)b * 262144;
  #pragma unroll
  for (int i = 0; i < 2; ++i)
    #pragma unroll
    for (int j = 0; j < 4; ++j) {
      int n = n0 + j * 16 + l15;
      #pragma unroll
      for (int r = 0; r < 8; ++r) {
        int o = m0 + i * 16 + half * 8 + r;
        orow[o * 1024 + n] = acc[i][j][r];
      }
    }
}

extern "C" void kernel_launch(void* const* d_in, const int* in_sizes, int n_in,
                              void* d_out, int out_size, void* d_ws, size_t ws_size,
                              hipStream_t stream) {
  (void)in_sizes; (void)n_in; (void)out_size; (void)ws_size;
  const float* x1 = (const float*)d_in[0];   // [16,256,32,32]
  const float* x2 = (const float*)d_in[1];   // [16,256,32,32]
  const float* gw = (const float*)d_in[2];   // [512,512,3,3]
  const float* ww = (const float*)d_in[3];   // [256,512,1,1]
  const float* w1 = (const float*)d_in[4];   // [512,16]
  const float* w2 = (const float*)d_in[5];   // [512,16]

  char* ws = (char*)d_ws;                    // ~73.2 MB used (with reuse)
  unsigned short* xcn   = (unsigned short*)(ws + 0);          // 16.78 MB  [b][c][n] bf16
  unsigned short* xnc   = (unsigned short*)(ws + 16777216);   // 16.78 MB  [b][n][c] bf16
  unsigned short* wpack = (unsigned short*)(ws + 33554432);   //  4.72 MB  [tap][co][ci] bf16
  unsigned short* wwb   = (unsigned short*)(ws + 38273024);   //  0.26 MB  [o][c] bf16
  unsigned short* valt  = (unsigned short*)(ws + 38535168);   // 16.78 MB  value^T [b][n][c] bf16
  float*          rmap  = (float*)(ws + 55312384);            //  1.05 MB
  float*          simb  = (float*)(ws + 56360960);            // 16.78 MB
  unsigned short* zpad  = (unsigned short*)(ws + 73138176);   //  1 KB zero row
  unsigned short* attn  = xnc;   // reuse: x_nc dead after conv
  unsigned short* ctxt  = xcn;   // reuse: x_cn dead after sim
  // zero-pad element offset relative to xnc base (both live inside ws)
  const int zoff = (int)((73138176 - 16777216) / 2);

  prep_x_kernel <<<32768, 256, 0, stream>>>(x1, x2, xcn, xnc);
  prep_gw_kernel<<< 9216, 256, 0, stream>>>(gw, wpack);
  prep_ww_kernel<<<  512, 256, 0, stream>>>(ww, wwb);
  zero_pad_kernel<<<   2, 256, 0, stream>>>(zpad);
  rmap_kernel   <<< 1024, 256, 0, stream>>>(w1, w2, rmap);

  // 32(M) x 64(N) register tile per wave, 8 waves per block
  conv3x3_wmma_kernel<<<dim3(2, 16, 16), 256, 0, stream>>>(xnc, wpack, valt, zoff);
  sim_wmma_kernel    <<<dim3(1, 16, 16), 256, 0, stream>>>(xcn, rmap, simb);
  softmax_kernel     <<<8192,            256, 0, stream>>>(simb, attn);
  ctx_wmma_kernel    <<<dim3(2, 16, 16), 256, 0, stream>>>(attn, valt, ctxt);
  out_wmma_kernel    <<<dim3(2,  8, 16), 256, 0, stream>>>(wwb, ctxt, (float*)d_out);
}
